// DDECModel_61692910240221
// MI455X (gfx1250) — matrix operand order repaired and checked by reading, not compiled, not run on hardware
//
#include <hip/hip_runtime.h>
#include <hip/hip_bf16.h>
#include <stddef.h>

typedef __attribute__((ext_vector_type(16))) __bf16 v16bf;
typedef __attribute__((ext_vector_type(8)))  float  v8f;
typedef __attribute__((ext_vector_type(4)))  unsigned int v4u;
typedef __attribute__((ext_vector_type(8)))  int    v8i;
typedef __attribute__((ext_vector_type(4)))  int    v4i;

#define EPS_C     0.01f
#define REGK_C    1e-6f
#define CG_ITERS  64
#define HW        5

#if __has_builtin(__builtin_amdgcn_tensor_load_to_lds) && __has_builtin(__builtin_amdgcn_s_wait_tensorcnt)
#define USE_TDM 1
#else
#define USE_TDM 0
#endif

// ---------- helpers ----------
__device__ inline unsigned short f2bf(float x) {
  union { float f; unsigned u; } v; v.f = x;
  unsigned r = v.u + 0x7FFFu + ((v.u >> 16) & 1u);   // round-to-nearest-even
  return (unsigned short)(r >> 16);
}
__device__ inline float bf2f(unsigned short h) {
  union { unsigned u; float f; } v; v.u = ((unsigned)h) << 16;
  return v.f;
}
__device__ inline float wave_reduce(float s) {
  #pragma unroll
  for (int off = 16; off > 0; off >>= 1) s += __shfl_down(s, off, 32);
  return s;
}

#if USE_TDM
// Issue one TDM 2D tile load: tile_w x tile_h fp32 elements, row stride
// `stride_elems`, packed contiguously into LDS at byte offset lds_off.
// Descriptor bitfields per CDNA5 ISA section 8.3/8.4 (D# group0/group1).
// This toolchain's builtin takes 6 args (g0, g1, g2, g3, g4, cpol).
__device__ inline void tdm_load_tile_f32(unsigned lds_off, const float* gptr,
                                         unsigned tile_w, unsigned tile_h,
                                         unsigned stride_elems) {
  unsigned long long ga = (unsigned long long)gptr;
  v4u g0;
  g0[0] = 1u;                                          // count=1, user descriptor
  g0[1] = lds_off;                                     // lds_addr (bytes)
  g0[2] = (unsigned)(ga & 0xFFFFFFFFull);              // global_addr[31:0]
  g0[3] = (unsigned)((ga >> 32) & 0x1FFFFFFull)        // global_addr[56:32]
        | (2u << 30);                                  // type=2 ("image")
  unsigned td0 = stride_elems;                         // tensor_dim0
  unsigned td1 = 1u << 20;                             // tensor_dim1 (ample)
  v8i g1;
  g1[0] = (int)(2u << 16);                             // data_size=2 (4 bytes)
  g1[1] = (int)((td0 & 0xFFFFu) << 16);                // tensor_dim0[15:0]
  g1[2] = (int)(((td0 >> 16) & 0xFFFFu) | ((td1 & 0xFFFFu) << 16));
  g1[3] = (int)(((td1 >> 16) & 0xFFFFu) | (tile_w << 16));   // tile_dim0
  g1[4] = (int)(tile_h & 0xFFFFu);                     // tile_dim1 (tile_dim2=0)
  g1[5] = (int)stride_elems;                           // tensor_dim0_stride[31:0]
  g1[6] = 0;                                           // stride hi / dim1_stride lo
  g1[7] = 0;                                           // dim1_stride hi
  v4i z4 = {0, 0, 0, 0};
  v8i z8 = {0, 0, 0, 0, 0, 0, 0, 0};
  __builtin_amdgcn_tensor_load_to_lds(g0, g1, z4, z4, z8, 0);
}
#endif

// ---------- prep: diagonal Hodge inverses, scalings, first RHS ----------
__global__ void prep_k(const float* __restrict__ B1, const float* __restrict__ B2,
                       const float* __restrict__ D1, const float* __restrict__ D2,
                       const float* __restrict__ f,
                       float* __restrict__ wv, float* __restrict__ B1inv,
                       float* __restrict__ D1inv, float* __restrict__ B2sq,
                       float* __restrict__ D2sq, float* __restrict__ dreg,
                       float* __restrict__ rhs0, int Mdim, int Ndim) {
  int i = blockIdx.x * blockDim.x + threadIdx.x;
  if (i < Ndim) {
    float b = B1[i], d = D1[i];
    float bi = 1.0f / (b * b + 1e-5f);
    float di = 1.0f / (d * d + 1e-5f);
    B1inv[i] = bi; D1inv[i] = di; wv[i] = bi * di;
  }
  if (i < Mdim) {
    float b = B2[i], d = D2[i];
    float bs = b * b, ds = d * d;
    B2sq[i] = bs; D2sq[i] = ds;
    dreg[i] = REGK_C / (bs * ds);
    rhs0[i] = f[i] / bs;
  }
}

// ---------- transpose d1 (M x N) -> d1T (N x M) ----------
__global__ void transpose_k(const float* __restrict__ in, float* __restrict__ out,
                            int rows, int cols) {
  __shared__ float t[32][33];
  int bx = blockIdx.x * 32, by = blockIdx.y * 32;
  #pragma unroll
  for (int j = 0; j < 32; j += 8) {
    int r = by + threadIdx.y + j, c = bx + threadIdx.x;
    if (r < rows && c < cols) t[threadIdx.y + j][threadIdx.x] = in[(size_t)r * cols + c];
  }
  __syncthreads();
  #pragma unroll
  for (int j = 0; j < 32; j += 8) {
    int c = bx + threadIdx.y + j, r = by + threadIdx.x;
    if (r < rows && c < cols) out[(size_t)c * rows + r] = t[threadIdx.x][threadIdx.y + j];
  }
}

// ---------- WMMA GEMM: S = A diag(w) A^T (+ dreg on the diagonal) ----------
// bf16x3 split (hi*hi + hi*lo + lo*hi) on v_wmma_f32_16x16x32_bf16.
// Tiles staged global->LDS by the Tensor Data Mover with double buffering so
// the DMA for tile t+1 overlaps the convert+WMMA of tile t, then split and
// swizzled into the ISA fragment layout (one 32B contiguous LDS row per lane).
__global__ __launch_bounds__(256) void gemm_awat_wmma(
    const float* __restrict__ A, const float* __restrict__ wv,
    const float* __restrict__ dreg, float* __restrict__ S,
    int Mdim, int Ndim) {
  __shared__ v16bf Ahi[4][32], Alo[4][32], Bhi[4][32], Blo[4][32];
#if USE_TDM
  __shared__ float stgA[2][64 * 32];
  __shared__ float stgB[2][64 * 32];
#endif
  const int tid  = threadIdx.x;
  const int lane = tid & 31;
  const int wave = tid >> 5;
  const int wr   = wave >> 2;       // 0..1
  const int wc   = wave & 3;        // 0..3
  const long bi  = (long)blockIdx.y * 64;
  const long bj  = (long)blockIdx.x * 64;

  v8f acc0 = {0.f,0.f,0.f,0.f,0.f,0.f,0.f,0.f};
  v8f acc1 = acc0;

  const int nkt = Ndim / 32;
#if USE_TDM
  if (wave == 0) {   // prologue: kick off tile 0
    tdm_load_tile_f32((unsigned)(unsigned long long)(&stgA[0][0]),
                      A + (size_t)bi * Ndim, 32u, 64u, (unsigned)Ndim);
    tdm_load_tile_f32((unsigned)(unsigned long long)(&stgB[0][0]),
                      A + (size_t)bj * Ndim, 32u, 64u, (unsigned)Ndim);
  }
#endif
  for (int t = 0; t < nkt; ++t) {
    const int kt = t * 32;
    __syncthreads();   // frags consumed; staging buffer (t+1)&1 free for refill
#if USE_TDM
    if (wave == 0) {
      if (t + 1 < nkt) {
        const int nb = (t + 1) & 1;
        tdm_load_tile_f32((unsigned)(unsigned long long)(&stgA[nb][0]),
                          A + (size_t)bi * Ndim + (kt + 32), 32u, 64u, (unsigned)Ndim);
        tdm_load_tile_f32((unsigned)(unsigned long long)(&stgB[nb][0]),
                          A + (size_t)bj * Ndim + (kt + 32), 32u, 64u, (unsigned)Ndim);
        __builtin_amdgcn_s_wait_tensorcnt(2);   // tile t's pair complete; t+1 in flight
      } else {
        __builtin_amdgcn_s_wait_tensorcnt(0);   // drain last tile
      }
    }
    __syncthreads();                 // tile t staging visible to all waves
    const float* sA = &stgA[t & 1][0];
    const float* sB = &stgB[t & 1][0];
#endif
    for (int e = tid; e < 64 * 32; e += 256) {
      int m = e >> 5;          // 0..63 row in tile
      int k = e & 31;          // 0..31 k in tile
#if USE_TDM
      float av = sA[e];
      float bv = sB[e] * wv[kt + k];
#else
      float av = A[(size_t)(bi + m) * Ndim + (kt + k)];
      float bv = A[(size_t)(bj + m) * Ndim + (kt + k)] * wv[kt + k];
#endif
      unsigned short ah = f2bf(av);
      unsigned short al = f2bf(av - bf2f(ah));
      unsigned short bh = f2bf(bv);
      unsigned short bl = f2bf(bv - bf2f(bh));
      int st = m >> 4, ml = m & 15;
      // A operand 16x32: lanes 0-15 hold K{0..7,16..23}, lanes 16-31 K{8..15,24..31}
      int laneA = ml + (((k >> 3) & 1) << 4);
      int idxA  = (k & 7) + ((k & 16) >> 1);
      ((unsigned short*)&Ahi[st][laneA])[idxA] = ah;
      ((unsigned short*)&Alo[st][laneA])[idxA] = al;
      // B operand 32x16: lanes 0-15 hold K{0..7,16..23}, lanes 16-31 K{8..15,24..31}
      int laneB = ml + (((k >> 3) & 1) << 4);
      int idxB  = (k & 7) + ((k & 16) >> 1);
      ((unsigned short*)&Bhi[st][laneB])[idxB] = bh;
      ((unsigned short*)&Blo[st][laneB])[idxB] = bl;
    }
    __syncthreads();
    v16bf bfh = Bhi[wc][lane];
    v16bf bfl = Blo[wc][lane];
    v16bf a0h = Ahi[2 * wr + 0][lane];
    v16bf a0l = Alo[2 * wr + 0][lane];
    v16bf a1h = Ahi[2 * wr + 1][lane];
    v16bf a1l = Alo[2 * wr + 1][lane];
    acc0 = __builtin_amdgcn_wmma_f32_16x16x32_bf16(false, a0h, false, bfh, (short)0, acc0, false, false);
    acc0 = __builtin_amdgcn_wmma_f32_16x16x32_bf16(false, a0h, false, bfl, (short)0, acc0, false, false);
    acc0 = __builtin_amdgcn_wmma_f32_16x16x32_bf16(false, a0l, false, bfh, (short)0, acc0, false, false);
    acc1 = __builtin_amdgcn_wmma_f32_16x16x32_bf16(false, a1h, false, bfh, (short)0, acc1, false, false);
    acc1 = __builtin_amdgcn_wmma_f32_16x16x32_bf16(false, a1h, false, bfl, (short)0, acc1, false, false);
    acc1 = __builtin_amdgcn_wmma_f32_16x16x32_bf16(false, a1l, false, bfh, (short)0, acc1, false, false);
  }

  // epilogue: C layout — VGPR r: lanes 0-15 -> M=r, lanes 16-31 -> M=r+8; N = lane&15
  const long col   = bj + wc * 16 + (lane & 15);
  // branchless diagonal regularization: col is fixed per lane, hoist the load
  const float dcol = (blockIdx.x == blockIdx.y) ? dreg[col] : 0.f;
  const int  rbase = (lane >> 4) << 3;
  const long row0  = bi + (2 * wr + 0) * 16 + rbase;
  const long row1  = bi + (2 * wr + 1) * 16 + rbase;
  #pragma unroll
  for (int r = 0; r < 8; ++r) {
    long rr0 = row0 + r;
    S[(size_t)rr0 * Mdim + col] = acc0[r] + ((rr0 == col) ? dcol : 0.f);
    long rr1 = row1 + r;
    S[(size_t)rr1 * Mdim + col] = acc1[r] + ((rr1 == col) ? dcol : 0.f);
  }
}

// ---------- wave-per-row matvec: out[i] = rs[i] * dot(mat[i,:], vec) ----------
__global__ void rowdot_k(const float* __restrict__ mat, const float* __restrict__ vec,
                         const float* __restrict__ rs, float* __restrict__ out,
                         int rows, int cols) {
  int row  = blockIdx.x * (blockDim.x >> 5) + (threadIdx.x >> 5);
  int lane = threadIdx.x & 31;
  if (row >= rows) return;
  const float4* rp = (const float4*)(mat + (size_t)row * cols);
  const float4* vp = (const float4*)vec;
  int c4 = cols >> 2;
  float s = 0.f;
  for (int c = lane; c < c4; c += 32) {
    __builtin_prefetch(rp + c + 128, 0, 1);
    float4 a = rp[c], b = vp[c];
    s += a.x * b.x + a.y * b.y + a.z * b.z + a.w * b.w;
  }
  s = wave_reduce(s);
  if (lane == 0) out[row] = (rs ? rs[row] : 1.0f) * s;
}

// ---------- elementwise scalar MLP (H=5) + analytic derivative ----------
__global__ void mlp_k(const float* __restrict__ g,
                      const float* __restrict__ W1, const float* __restrict__ b1,
                      const float* __restrict__ W2, const float* __restrict__ b2,
                      const float* __restrict__ W3, const float* __restrict__ b3,
                      const float* __restrict__ W4, const float* __restrict__ b4,
                      const float* __restrict__ B1inv, const float* __restrict__ wv,
                      float* __restrict__ zb, float* __restrict__ wj, int Ndim) {
  int n = blockIdx.x * blockDim.x + threadIdx.x;
  if (n >= Ndim) return;
  float x = g[n];
  float h1[HW], t1[HW], h2[HW], t2[HW], h3[HW], t3[HW];
  #pragma unroll
  for (int j = 0; j < HW; ++j) {
    float th = tanhf(W1[j] * x + b1[j]);
    h1[j] = th; t1[j] = (1.f - th * th) * W1[j];
  }
  #pragma unroll
  for (int j = 0; j < HW; ++j) {
    float pre = b2[j], dt = 0.f;
    #pragma unroll
    for (int k = 0; k < HW; ++k) { pre += W2[j * HW + k] * h1[k]; dt += W2[j * HW + k] * t1[k]; }
    float th = tanhf(pre);
    h2[j] = th; t2[j] = (1.f - th * th) * dt;
  }
  #pragma unroll
  for (int j = 0; j < HW; ++j) {
    float pre = b3[j], dt = 0.f;
    #pragma unroll
    for (int k = 0; k < HW; ++k) { pre += W3[j * HW + k] * h2[k]; dt += W3[j * HW + k] * t2[k]; }
    float th = tanhf(pre);
    h3[j] = th; t3[j] = (1.f - th * th) * dt;
  }
  float out = b4[0], dout = 0.f;
  #pragma unroll
  for (int k = 0; k < HW; ++k) { out += W4[k] * h3[k]; dout += W4[k] * t3[k]; }
  zb[n] = B1inv[n] * out;                  // B1^-1 * mlp(g)
  wj[n] = wv[n] * (1.f + EPS_C * dout);    // Jacobian GEMM weight
}

// ---------- Newton residual (y-space RHS): rhs = (Ku + eps*DIVz - 2f)/B2^2 ----------
__global__ void residual_k(const float* __restrict__ Kt, const float* __restrict__ u,
                           const float* __restrict__ divz, const float* __restrict__ f,
                           const float* __restrict__ B2sq, float* __restrict__ rhs,
                           int Mdim) {
  int i = blockIdx.x * blockDim.x + threadIdx.x;
  if (i >= Mdim) return;
  float Ku = Kt[i] + REGK_C * u[i];
  float rx = Ku + EPS_C * divz[i] - 2.f * f[i];
  rhs[i] = rx / B2sq[i];
}

// ---------- tiny elementwise kernels ----------
__global__ void vmul_k(float* __restrict__ out, const float* __restrict__ a,
                       const float* __restrict__ b, int n) {
  int i = blockIdx.x * blockDim.x + threadIdx.x;
  if (i < n) out[i] = a[i] * b[i];
}
__global__ void vdiv_k(float* __restrict__ out, const float* __restrict__ a,
                       const float* __restrict__ b, int n) {
  int i = blockIdx.x * blockDim.x + threadIdx.x;
  if (i < n) out[i] = a[i] / b[i];
}
__global__ void vsubdiv_k(float* __restrict__ u, const float* __restrict__ x,
                          const float* __restrict__ b, int n) {
  int i = blockIdx.x * blockDim.x + threadIdx.x;
  if (i < n) u[i] -= x[i] / b[i];
}
__global__ void copy_k(float* __restrict__ out, const float* __restrict__ in, int n) {
  int i = blockIdx.x * blockDim.x + threadIdx.x;
  if (i < n) out[i] = in[i];
}

// ---------- CG on the symmetrized SPD system ----------
__global__ void cg_init_k(float* __restrict__ x, float* __restrict__ r,
                          float* __restrict__ p, const float* __restrict__ rhs,
                          float* __restrict__ sc, int n) {
  __shared__ float red[256];
  float acc = 0.f;
  for (int i = threadIdx.x; i < n; i += blockDim.x) {
    float v = rhs[i];
    x[i] = 0.f; r[i] = v; p[i] = v;
    acc += v * v;
  }
  red[threadIdx.x] = acc; __syncthreads();
  for (int s = 128; s > 0; s >>= 1) {
    if ((int)threadIdx.x < s) red[threadIdx.x] += red[threadIdx.x + s];
    __syncthreads();
  }
  if (threadIdx.x == 0) { sc[0] = red[0]; sc[1] = 0.f; }
}

__global__ void cg_spmv_k(const float* __restrict__ Mat, const float* __restrict__ p,
                          float* __restrict__ y, float* __restrict__ sc, int n) {
  int row  = blockIdx.x * (blockDim.x >> 5) + (threadIdx.x >> 5);
  int lane = threadIdx.x & 31;
  if (row >= n) return;
  const float4* rp = (const float4*)(Mat + (size_t)row * n);
  const float4* pp = (const float4*)p;
  int c4 = n >> 2;
  float s = 0.f;
  for (int c = lane; c < c4; c += 32) {
    __builtin_prefetch(rp + c + 128, 0, 1);
    float4 a = rp[c], b = pp[c];
    s += a.x * b.x + a.y * b.y + a.z * b.z + a.w * b.w;
  }
  s = wave_reduce(s);
  if (lane == 0) {
    y[row] = s;
    atomicAdd(&sc[1], p[row] * s);   // p^T A p partial
  }
}

__global__ void cg_update_k(float* __restrict__ x, float* __restrict__ r,
                            float* __restrict__ p, const float* __restrict__ y,
                            float* __restrict__ sc, int n) {
  __shared__ float red[256];
  __shared__ float sh[2];
  if (threadIdx.x == 0) {
    float rr = sc[0], pAp = sc[1];
    sh[0] = (pAp != 0.f) ? rr / pAp : 0.f;   // alpha
    sh[1] = rr;
  }
  __syncthreads();
  float alpha = sh[0];
  float acc = 0.f;
  for (int i = threadIdx.x; i < n; i += blockDim.x) {
    x[i] += alpha * p[i];
    float ri = r[i] - alpha * y[i];
    r[i] = ri;
    acc += ri * ri;
  }
  red[threadIdx.x] = acc; __syncthreads();
  for (int s = 128; s > 0; s >>= 1) {
    if ((int)threadIdx.x < s) red[threadIdx.x] += red[threadIdx.x + s];
    __syncthreads();
  }
  float beta = (sh[1] != 0.f) ? red[0] / sh[1] : 0.f;
  for (int i = threadIdx.x; i < n; i += blockDim.x) p[i] = r[i] + beta * p[i];
  if (threadIdx.x == 0) { sc[0] = red[0]; sc[1] = 0.f; }
}

// ---------- host-side orchestration ----------
static void run_cg(const float* Mat, const float* rhs,
                   float* xv, float* rv, float* pv, float* yv, float* sc,
                   int n, hipStream_t stream) {
  cg_init_k<<<1, 256, 0, stream>>>(xv, rv, pv, rhs, sc, n);
  int spmv_blocks = (n + 7) / 8;
  for (int it = 0; it < CG_ITERS; ++it) {
    cg_spmv_k<<<spmv_blocks, 256, 0, stream>>>(Mat, pv, yv, sc, n);
    cg_update_k<<<1, 256, 0, stream>>>(xv, rv, pv, yv, sc, n);
  }
}

extern "C" void kernel_launch(void* const* d_in, const int* in_sizes, int n_in,
                              void* d_out, int out_size, void* d_ws, size_t ws_size,
                              hipStream_t stream) {
  const int Mdim = in_sizes[0];    // 2048 (u)
  const int Ndim = in_sizes[4];    // 4096 (B1_vals)

  const float* f_in = (const float*)d_in[1];
  const float* d1   = (const float*)d_in[3];
  const float* B1v  = (const float*)d_in[4];
  const float* B2v  = (const float*)d_in[5];
  const float* D1v  = (const float*)d_in[6];
  const float* D2v  = (const float*)d_in[7];
  const float* W1   = (const float*)d_in[8];
  const float* b1   = (const float*)d_in[9];
  const float* W2   = (const float*)d_in[10];
  const float* b2   = (const float*)d_in[11];
  const float* W3   = (const float*)d_in[12];
  const float* b3   = (const float*)d_in[13];
  const float* W4   = (const float*)d_in[14];
  const float* b4   = (const float*)d_in[15];
  float* out = (float*)d_out;

  float* W  = (float*)d_ws;
  size_t o = 0;
  float* d1T   = W + o; o += (size_t)Ndim * Mdim;
  float* S_K   = W + o; o += (size_t)Mdim * Mdim;
  float* S_J   = W + o; o += (size_t)Mdim * Mdim;
  float* wv    = W + o; o += Ndim;
  float* B1inv = W + o; o += Ndim;
  float* D1inv = W + o; o += Ndim;
  float* wj    = W + o; o += Ndim;
  float* gvec  = W + o; o += Ndim;
  float* zb    = W + o; o += Ndim;
  float* B2sq  = W + o; o += Mdim;
  float* D2sq  = W + o; o += Mdim;
  float* dreg  = W + o; o += Mdim;
  float* rhs0  = W + o; o += Mdim;
  float* uvec  = W + o; o += Mdim;
  float* ud    = W + o; o += Mdim;
  float* divz  = W + o; o += Mdim;
  float* Kt    = W + o; o += Mdim;
  float* rhs   = W + o; o += Mdim;
  float* xv    = W + o; o += Mdim;
  float* rv    = W + o; o += Mdim;
  float* pv    = W + o; o += Mdim;
  float* yv    = W + o; o += Mdim;
  float* sc    = W + o; o += 8;
  (void)ws_size; (void)n_in;

  const int TPB = 256;
  int gN = (Ndim + TPB - 1) / TPB;
  int gM = (Mdim + TPB - 1) / TPB;

  prep_k<<<gN, TPB, 0, stream>>>(B1v, B2v, D1v, D2v, f_in,
                                 wv, B1inv, D1inv, B2sq, D2sq, dreg, rhs0, Mdim, Ndim);
  transpose_k<<<dim3((Ndim + 31) / 32, (Mdim + 31) / 32), dim3(32, 8), 0, stream>>>(d1, d1T, Mdim, Ndim);

  dim3 ggrid(Mdim / 64, Mdim / 64);
  gemm_awat_wmma<<<ggrid, 256, 0, stream>>>(d1, wv, dreg, S_K, Mdim, Ndim);
  run_cg(S_K, rhs0, xv, rv, pv, yv, sc, Mdim, stream);
  vdiv_k<<<gM, TPB, 0, stream>>>(uvec, xv, D2sq, Mdim);

  for (int it = 0; it < 3; ++it) {
    vmul_k<<<gM, TPB, 0, stream>>>(ud, D2sq, uvec, Mdim);
    rowdot_k<<<(Ndim + 7) / 8, 256, 0, stream>>>(d1T, ud, D1inv, gvec, Ndim, Mdim);
    mlp_k<<<gN, TPB, 0, stream>>>(gvec, W1, b1, W2, b2, W3, b3, W4, b4,
                                  B1inv, wv, zb, wj, Ndim);
    rowdot_k<<<(Mdim + 7) / 8, 256, 0, stream>>>(d1, zb, B2sq, divz, Mdim, Ndim);
    rowdot_k<<<(Mdim + 7) / 8, 256, 0, stream>>>(S_K, ud, B2sq, Kt, Mdim, Mdim);
    residual_k<<<gM, TPB, 0, stream>>>(Kt, uvec, divz, f_in, B2sq, rhs, Mdim);
    gemm_awat_wmma<<<ggrid, 256, 0, stream>>>(d1, wj, dreg, S_J, Mdim, Ndim);
    run_cg(S_J, rhs, xv, rv, pv, yv, sc, Mdim, stream);
    vsubdiv_k<<<gM, TPB, 0, stream>>>(uvec, xv, D2sq, Mdim);
  }

  copy_k<<<(out_size + TPB - 1) / TPB, TPB, 0, stream>>>(out, uvec, out_size);
}
